// BiMiniGRU_3650722201915
// MI455X (gfx1250) — matrix operand order repaired and checked by reading, not compiled, not run on hardware
//
#include <hip/hip_runtime.h>
#include <hip/hip_bf16.h>

#define B_    8
#define L_    8192
#define C_    256
#define D_    256
#define M_    (B_ * L_)        // 65536 rows
#define NSEG  6
#define NCAT  (NSEG * D_)      // 1536 concatenated output cols
#define CHUNK 128
#define NCH   (L_ / CHUNK)     // 64 chunks per sequence

typedef __attribute__((ext_vector_type(16))) __bf16 bf16x16;
typedef __attribute__((ext_vector_type(8)))  __bf16 bf16x8;
typedef __attribute__((ext_vector_type(8)))  float  f32x8;

// Fast sigmoid: v_exp_f32 + v_rcp_f32 (no IEEE div sequence); precision is
// already bounded by the bf16 GEMM inputs.
__device__ __forceinline__ float sigmoidf_(float x) {
    return __builtin_amdgcn_rcpf(1.0f + __expf(-x));
}

// ---------------------------------------------------------------------------
// Prep: pack the six [C,D] fp32 weights into Wt bf16, layout wt[n_cat*256 + k]
// = W_seg[k, n] (k-contiguous per output column -> contiguous 32B B-operand
// loads for v_wmma). Also concatenate the six biases.
// ---------------------------------------------------------------------------
__global__ __launch_bounds__(256) void prep_kernel(
    const float* __restrict__ w0, const float* __restrict__ w1,
    const float* __restrict__ w2, const float* __restrict__ w3,
    const float* __restrict__ w4, const float* __restrict__ w5,
    const float* __restrict__ b0, const float* __restrict__ b1,
    const float* __restrict__ b2, const float* __restrict__ b3,
    const float* __restrict__ b4, const float* __restrict__ b5,
    __bf16* __restrict__ wt, float* __restrict__ bcat)
{
    const int gid = blockIdx.x * 256 + threadIdx.x;   // 0 .. 6*256*256-1
    const int k   = gid & 255;
    const int ng  = gid >> 8;                         // 0..1535
    const int seg = ng >> 8;
    const int n   = ng & 255;
    const float* W = w0;
    switch (seg) { case 1: W = w1; break; case 2: W = w2; break;
                   case 3: W = w3; break; case 4: W = w4; break;
                   case 5: W = w5; break; default: break; }
    wt[(size_t)ng * C_ + k] = (__bf16)W[k * D_ + n];
    if (gid < NCAT) {
        const int s2 = gid >> 8, n2 = gid & 255;
        const float* bb = b0;
        switch (s2) { case 1: bb = b1; break; case 2: bb = b2; break;
                      case 3: bb = b3; break; case 4: bb = b4; break;
                      case 5: bb = b5; break; default: break; }
        bcat[gid] = bb[n2];
    }
}

// ---------------------------------------------------------------------------
// Fused 6-way GEMM: Y[m, 0:1536] = xs[m,:] @ [Wh1|Wz1|Ws1|Wh_1|Wz_1|Ws_1] + b
// Block = 256 threads = 8 waves; 64 rows of xs staged once in LDS as bf16.
// B-stationary: each wave owns 12 distinct col-tiles (nt = w, w+8, ...);
// per tile it loads the 8 B operands (K=256) into registers ONCE, then
// sweeps the 4 m-tiles (A operands register-cached from LDS, hoisted by the
// compiler out of the nt loop) -> inner loop is 8 back-to-back v_wmma.
// Each B tile is read from L2 exactly once per block.
// ---------------------------------------------------------------------------
__global__ __launch_bounds__(256) void gemm_kernel(
    const float* __restrict__ xs, const __bf16* __restrict__ wt,
    const float* __restrict__ bcat,
    float* __restrict__ g0, float* __restrict__ g1, float* __restrict__ g2,
    float* __restrict__ g3, float* __restrict__ g4, float* __restrict__ g5)
{
    __shared__ __bf16 lx[64][264];                  // pad 8 -> 16B-aligned runs
    const int m0 = blockIdx.x * 64;
    const int t  = threadIdx.x;
    #pragma unroll
    for (int i = 0; i < 64; ++i)
        lx[i][t] = (__bf16)xs[(size_t)(m0 + i) * C_ + t];
    __syncthreads();

    const int w    = t >> 5;
    const int lane = t & 31;
    const int l15  = lane & 15;
    const int hi16 = lane & 16;                     // 0 or 16
    const int kbA  = hi16 >> 1;                     // A: K-half base 0/16 (used as 2*kbA)

    for (int nt = w; nt < NCAT / 16; nt += 8) {
        const int ng  = nt * 16 + l15;              // global col
        const int seg = nt >> 4;                    // uniform per wave
        const int d   = ng & 255;
        const __bf16* wp = wt + (size_t)ng * C_ + hi16;   // B: K-half base 0/16

        // Load this column tile's entire K=256 B operand into registers.
        bf16x16 breg[8];
        #pragma unroll
        for (int kk = 0; kk < 8; ++kk)
            breg[kk] = *(const bf16x16*)(wp + kk * 32);    // 32B contiguous

        const float bias = bcat[ng];
        float* outp = g0;
        switch (seg) { case 1: outp = g1; break; case 2: outp = g2; break;
                       case 3: outp = g3; break; case 4: outp = g4; break;
                       case 5: outp = g5; break; default: break; }
        const bool act = (seg != 0 && seg != 3);    // z and s columns

        #pragma unroll
        for (int mt = 0; mt < 4; ++mt) {
            const int arow = mt * 16 + l15;         // A: row = lane&15
            f32x8 acc;
            #pragma unroll
            for (int i = 0; i < 8; ++i) acc[i] = bias;
            #pragma unroll
            for (int kk = 0; kk < 8; ++kk) {
                bf16x8 lo = *(const bf16x8*)&lx[arow][kk * 32 + 2 * kbA];
                bf16x8 hh = *(const bf16x8*)&lx[arow][kk * 32 + 2 * kbA + 8];
                bf16x16 a;
                #pragma unroll
                for (int j = 0; j < 8; ++j) { a[j] = lo[j]; a[8 + j] = hh[j]; }
                acc = __builtin_amdgcn_wmma_f32_16x16x32_bf16(
                    false, a, false, breg[kk], (short)0, acc, false, false);
            }
            if (act) {
                #pragma unroll
                for (int i = 0; i < 8; ++i) acc[i] = sigmoidf_(acc[i]);
            }
            const int mb = m0 + mt * 16 + (hi16 >> 1);   // D: row = i + 0/8
            #pragma unroll
            for (int i = 0; i < 8; ++i)
                outp[(size_t)(mb + i) * D_ + d] = acc[i];
        }
    }
}

// ---------------------------------------------------------------------------
// Scan phase A: per (dir, b, chunk, d) compose the 128 affine maps
// h -> (1-z)h + z*h~  into one (A, B) pair.  dir=1 walks time reversed.
// ---------------------------------------------------------------------------
__global__ __launch_bounds__(256) void scanA_kernel(
    const float* __restrict__ zf, const float* __restrict__ hf,
    const float* __restrict__ zb, const float* __restrict__ hb,
    float* __restrict__ Af, float* __restrict__ Bf,
    float* __restrict__ Ab, float* __restrict__ Bb)
{
    const int d = threadIdx.x, ch = blockIdx.x, b = blockIdx.y, dir = blockIdx.z;
    const float* Z = dir ? zb : zf;
    const float* H = dir ? hb : hf;
    float aacc = 1.0f, bacc = 0.0f;
    const size_t base = (size_t)b * L_ * D_ + d;
    #pragma unroll 4
    for (int i = 0; i < CHUNK; ++i) {
        const int r  = ch * CHUNK + i;
        const int tt = dir ? (L_ - 1 - r) : r;
        const size_t idx = base + (size_t)tt * D_;
        const float z = Z[idx], h = H[idx];
        const float a = 1.0f - z, bb = z * h;
        bacc = fmaf(a, bacc, bb);
        aacc *= a;
    }
    const size_t o = (size_t)(b * NCH + ch) * D_ + d;
    if (dir) { Ab[o] = aacc; Bb[o] = bacc; }
    else     { Af[o] = aacc; Bf[o] = bacc; }
}

// Phase B: serial scan over the 64 chunk transforms per (dir, b, d);
// overwrites A with the per-chunk carry-in h value.
__global__ __launch_bounds__(256) void scanB_kernel(
    float* __restrict__ Af, const float* __restrict__ Bf,
    float* __restrict__ Ab, const float* __restrict__ Bb,
    const float* __restrict__ h0f, const float* __restrict__ h0b)
{
    const int d = threadIdx.x, b = blockIdx.x, dir = blockIdx.y;
    float* A        = dir ? Ab : Af;
    const float* Bo = dir ? Bb : Bf;
    float h = dir ? h0b[d] : h0f[d];
    for (int c = 0; c < NCH; ++c) {
        const size_t o = (size_t)(b * NCH + c) * D_ + d;
        const float av = A[o], bv = Bo[o];
        A[o] = h;                    // carry-in for phase C
        h = fmaf(av, h, bv);
    }
}

// Phase C forward: replay each chunk with its carry-in, write hf * sigmoid(sf)
// in place over the h~ array.
__global__ __launch_bounds__(256) void scanCf_kernel(
    const float* __restrict__ zf, float* __restrict__ hf,
    const float* __restrict__ sf, const float* __restrict__ carry)
{
    const int d = threadIdx.x, ch = blockIdx.x, b = blockIdx.y;
    float h = carry[(size_t)(b * NCH + ch) * D_ + d];
    const size_t base = (size_t)b * L_ * D_ + d;
    #pragma unroll 4
    for (int i = 0; i < CHUNK; ++i) {
        const size_t idx = base + (size_t)(ch * CHUNK + i) * D_;
        const float z = zf[idx], ht = hf[idx];
        h = fmaf(1.0f - z, h, z * ht);
        hf[idx] = h * sf[idx];
    }
}

// Phase C backward + final combine: out = hb*sigmoid(sb) + (hf*sigmoid(sf)).
__global__ __launch_bounds__(256) void scanCb_kernel(
    const float* __restrict__ zb, const float* __restrict__ hb,
    const float* __restrict__ sb, const float* __restrict__ carry,
    const float* __restrict__ hfs, float* __restrict__ out)
{
    const int d = threadIdx.x, ch = blockIdx.x, b = blockIdx.y;
    float h = carry[(size_t)(b * NCH + ch) * D_ + d];
    const size_t base = (size_t)b * L_ * D_ + d;
    #pragma unroll 4
    for (int i = 0; i < CHUNK; ++i) {
        const int tt = L_ - 1 - (ch * CHUNK + i);
        const size_t idx = base + (size_t)tt * D_;
        const float z = zb[idx], ht = hb[idx];
        h = fmaf(1.0f - z, h, z * ht);
        out[idx] = fmaf(h, sb[idx], hfs[idx]);
    }
}

extern "C" void kernel_launch(void* const* d_in, const int* in_sizes, int n_in,
                              void* d_out, int out_size, void* d_ws, size_t ws_size,
                              hipStream_t stream) {
    (void)in_sizes; (void)n_in; (void)out_size; (void)ws_size;
    const float* xs   = (const float*)d_in[0];
    const float* Wh1  = (const float*)d_in[1];
    const float* bh1  = (const float*)d_in[2];
    const float* Wz1  = (const float*)d_in[3];
    const float* bz1  = (const float*)d_in[4];
    const float* Ws1  = (const float*)d_in[5];
    const float* bs1  = (const float*)d_in[6];
    const float* h01  = (const float*)d_in[7];
    const float* Wh_1 = (const float*)d_in[8];
    const float* bh_1 = (const float*)d_in[9];
    const float* Wz_1 = (const float*)d_in[10];
    const float* bz_1 = (const float*)d_in[11];
    const float* Ws_1 = (const float*)d_in[12];
    const float* bs_1 = (const float*)d_in[13];
    const float* h0_1 = (const float*)d_in[14];
    float* out = (float*)d_out;

    char* ws = (char*)d_ws;
    size_t off = 0;
    __bf16* wt  = (__bf16*)(ws + off); off += (size_t)NCAT * C_ * sizeof(__bf16);
    float* bcat = (float*)(ws + off);  off += (size_t)NCAT * sizeof(float);
    off = (off + 255) & ~(size_t)255;
    const size_t BLD = (size_t)M_ * D_;
    float* g[6];
    for (int i = 0; i < 6; ++i) { g[i] = (float*)(ws + off); off += BLD * sizeof(float); }
    const size_t CHN = (size_t)B_ * NCH * D_;
    float* Af = (float*)(ws + off); off += CHN * sizeof(float);
    float* Bf = (float*)(ws + off); off += CHN * sizeof(float);
    float* Ab = (float*)(ws + off); off += CHN * sizeof(float);
    float* Bb = (float*)(ws + off); off += CHN * sizeof(float);

    prep_kernel<<<(NSEG * C_ * D_) / 256, 256, 0, stream>>>(
        Wh1, Wz1, Ws1, Wh_1, Wz_1, Ws_1,
        bh1, bz1, bs1, bh_1, bz_1, bs_1, wt, bcat);

    gemm_kernel<<<M_ / 64, 256, 0, stream>>>(
        xs, wt, bcat, g[0], g[1], g[2], g[3], g[4], g[5]);

    scanA_kernel<<<dim3(NCH, B_, 2), 256, 0, stream>>>(
        g[1], g[0], g[4], g[3], Af, Bf, Ab, Bb);

    scanB_kernel<<<dim3(B_, 2), 256, 0, stream>>>(Af, Bf, Ab, Bb, h01, h0_1);

    scanCf_kernel<<<dim3(NCH, B_), 256, 0, stream>>>(g[1], g[0], g[2], Af);

    scanCb_kernel<<<dim3(NCH, B_), 256, 0, stream>>>(g[4], g[3], g[5], Ab, g[0], out);
}
